// LSTMBlock_13194139533701
// MI455X (gfx1250) — compile-verified, matrix operand order checked by reading, not run on hardware
//
#include <hip/hip_runtime.h>
#include <hip/hip_bf16.h>
#include <stdint.h>

// LSTM: T=512, B=64, IN=1024, H=1024. fp32 in/out, bf16 WMMA internally.
#define T_STEPS 512
#define B_SZ    64
#define IN_SZ   1024
#define H_SZ    1024
#define K_TOT   2048        // IN + H (fused [x;h] GEMM)
#define N_G     4096        // 4*H gate columns
#define KC      256         // K-chunk staged per TDM transfer
#define NCHUNK  (K_TOT / KC)

typedef __attribute__((ext_vector_type(16))) __bf16 v16bf;
typedef __attribute__((ext_vector_type(8)))  float  v8f;
typedef unsigned int u32;
typedef u32 u32x4 __attribute__((ext_vector_type(4)));
typedef u32 u32x8 __attribute__((ext_vector_type(8)));

__device__ __forceinline__ unsigned short f32_to_bf16_rn(float f) {
  union { float f; unsigned int u; } x; x.f = f;
  unsigned int u = x.u;
  unsigned int r = (u + 0x7FFFu + ((u >> 16) & 1u)) >> 16;
  return (unsigned short)r;
}

__device__ __forceinline__ float sigmoid_f(float v) {
  return 1.0f / (1.0f + __expf(-v));
}
__device__ __forceinline__ float tanh_f(float v) {
  return 2.0f / (1.0f + __expf(-2.0f * v)) - 1.0f;
}

// ---------------------------------------------------------------------------
// Phase-1 kernels
// ---------------------------------------------------------------------------

// Wt[n][k] = bf16( k<IN ? W_x[k][n] : W_h[k-IN][n] )  -- transposed, bf16
__global__ void convert_w_kernel(const float* __restrict__ Wx,
                                 const float* __restrict__ Wh,
                                 unsigned short* __restrict__ Wt) {
  size_t idx = (size_t)blockIdx.x * blockDim.x + threadIdx.x;
  if (idx >= (size_t)N_G * K_TOT) return;
  int n = (int)(idx >> 11);            // K_TOT = 2048 = 1<<11
  int k = (int)(idx & (K_TOT - 1));
  float v = (k < IN_SZ) ? Wx[(size_t)k * N_G + n]
                        : Wh[(size_t)(k - IN_SZ) * N_G + n];
  Wt[idx] = f32_to_bf16_rn(v);
}

__global__ void convert_x_kernel(const float* __restrict__ x,
                                 unsigned short* __restrict__ xbf,
                                 size_t n) {
  size_t idx = (size_t)blockIdx.x * blockDim.x + threadIdx.x;
  if (idx < n) {
    float v = __builtin_nontemporal_load(x + idx);
    xbf[idx] = f32_to_bf16_rn(v);
  }
}

__global__ void init_state_kernel(unsigned short* __restrict__ h0,
                                  float* __restrict__ c) {
  int idx = blockIdx.x * blockDim.x + threadIdx.x;
  if (idx < B_SZ * H_SZ) { h0[idx] = 0; c[idx] = 0.0f; }
}

// ---------------------------------------------------------------------------
// TDM: stage one 64-row x KC-col bf16 weight chunk into LDS.
// Tensor view of Wt [N_G rows, K_TOT cols] row-major, 3D tile:
//   dim0 (x): KC elements along K           (tensor_dim0 = K_TOT, stride row = K_TOT)
//   dim1 (y): 16 rows (cols of one gate)    (dim0_stride  = K_TOT)
//   dim2 (z): 4 gate slices                 (dim1_stride  = H_SZ*K_TOT)
// LDS result layout: [gate][row][k] contiguous, (4*16*KC) bf16.
// ---------------------------------------------------------------------------
__device__ __forceinline__ void tdm_load_w_chunk(const unsigned short* Wt,
                                                 int nBase, int chunk,
                                                 unsigned short* ldsDst) {
  const uint64_t ga =
      (uint64_t)(uintptr_t)(Wt + (size_t)nBase * K_TOT + (size_t)chunk * KC);
  const u32 lds = (u32)(uintptr_t)ldsDst;

  u32x4 g0;
  g0[0] = 1u;                                   // count=1, user descriptor
  g0[1] = lds;                                  // lds_addr
  g0[2] = (u32)(ga & 0xFFFFFFFFu);              // global_addr[31:0]
  g0[3] = (u32)((ga >> 32) & 0x01FFFFFFu)       // global_addr[56:32]
          | (2u << 30);                         // type = 2 ("image")

  u32x8 g1;
  g1[0] = (1u << 16);                           // data_size = 2 bytes
  g1[1] = ((u32)(K_TOT & 0xFFFF)) << 16;        // tensor_dim0[15:0]
  g1[2] = ((u32)(N_G & 0xFFFF)) << 16;          // dim0 hi=0 | tensor_dim1[15:0]
  g1[3] = ((u32)KC) << 16;                      // dim1 hi=0 | tile_dim0
  g1[4] = 16u | (4u << 16);                     // tile_dim1=16, tile_dim2=4
  g1[5] = (u32)K_TOT;                           // tensor_dim0_stride[31:0]
  g1[6] = 0u;                                   // stride0 hi | stride1[15:0]=0
  g1[7] = (u32)(((uint64_t)H_SZ * K_TOT) >> 16);// tensor_dim1_stride[47:16]

  u32x4 g2;
  g2[0] = 4u;                                   // tensor_dim2 = 4
  g2[1] = 0u; g2[2] = 0u; g2[3] = 0u;           // no dim3 / no iterate

  u32x4 g3;
  g3[0] = 0u; g3[1] = 0u; g3[2] = 0u; g3[3] = 0u;

  asm volatile("tensor_load_to_lds %0, %1, %2, %3"
               :: "s"(g0), "s"(g1), "s"(g2), "s"(g3)
               : "memory");
}

// ---------------------------------------------------------------------------
// One LSTM step: gates = [x_t ; h] @ Wt^T + b ; elementwise LSTM update.
// grid = 64 blocks (one per 16-wide h-column tile), block = 128 threads
// (4 waves = the 4 M-tiles covering B=64). Each wave computes four 16x16
// gate tiles (i,f,g,o) over K=2048 with v_wmma_f32_16x16x32_bf16, with
// weight chunks TDM-staged into double-buffered LDS shared by all 4 waves.
// ---------------------------------------------------------------------------
__global__ __launch_bounds__(128)
void lstm_step_kernel(const unsigned short* __restrict__ xbf,  // [T,B,IN] bf16
                      const unsigned short* __restrict__ Wt,   // [N_G,K_TOT] bf16
                      const float* __restrict__ bias,          // [N_G]
                      const unsigned short* __restrict__ hin,  // [B,H] bf16
                      unsigned short* __restrict__ hout,       // [B,H] bf16
                      float* __restrict__ c,                   // [B,H] f32
                      float* __restrict__ out,                 // hs | hT | cT
                      int t) {
  __shared__ __align__(16) unsigned short Bbuf[2][4 * 16 * KC];  // 2 x 32 KB

  const int lane  = (int)(threadIdx.x & 31u);
  const int wv    = (int)(threadIdx.x >> 5);   // 0..3 -> M tile
  const int nTile = (int)blockIdx.x;           // 0..63 -> h column tile
  const int mBase = wv * 16;
  const int nBase = nTile * 16;

  // ISA 16-bit fragment layout: lanes 0-15 hold K {0..7,16..23},
  // lanes 16-31 hold K {8..15,24..31}; two 8-elem (16B) chunks per lane.
  const int rowA  = mBase + (lane & 15);
  const int khalf = (lane < 16) ? 0 : 8;
  const int nLoc  = lane & 15;

  const unsigned short* xRow = xbf + ((size_t)t * B_SZ + rowA) * IN_SZ;
  const unsigned short* hRow = hin + (size_t)rowA * H_SZ;

  v8f acc0 = {}, acc1 = {}, acc2 = {}, acc3 = {};

  // Prologue: stage chunk 0.
  if (wv == 0) {
    tdm_load_w_chunk(Wt, nBase, 0, &Bbuf[0][0]);
    __builtin_amdgcn_s_wait_tensorcnt(0);
  }
  __syncthreads();

  for (int cchunk = 0; cchunk < NCHUNK; ++cchunk) {
    const int cur = cchunk & 1;
    // Kick off DMA for next chunk; overlaps with WMMA below.
    if (wv == 0 && (cchunk + 1) < NCHUNK) {
      tdm_load_w_chunk(Wt, nBase, cchunk + 1, &Bbuf[cur ^ 1][0]);
    }

    const unsigned short* Bc = &Bbuf[cur][0];
    const int kgBase = cchunk * KC;

#pragma unroll 4
    for (int kb = 0; kb < KC; kb += 32) {
      const int kg = kgBase + kb;
      const unsigned short* aSrc =
          (kg < IN_SZ) ? (xRow + kg) : (hRow + (kg - IN_SZ));
      const unsigned short* b0p = Bc + ((0 * 16 + nLoc) * KC + kb + khalf);
      const unsigned short* b1p = Bc + ((1 * 16 + nLoc) * KC + kb + khalf);
      const unsigned short* b2p = Bc + ((2 * 16 + nLoc) * KC + kb + khalf);
      const unsigned short* b3p = Bc + ((3 * 16 + nLoc) * KC + kb + khalf);

      v16bf a, b0, b1, b2, b3;
      ((uint4*)&a)[0]  = *(const uint4*)(aSrc + khalf);
      ((uint4*)&a)[1]  = *(const uint4*)(aSrc + khalf + 16);
      ((uint4*)&b0)[0] = *(const uint4*)(b0p);
      ((uint4*)&b0)[1] = *(const uint4*)(b0p + 16);
      ((uint4*)&b1)[0] = *(const uint4*)(b1p);
      ((uint4*)&b1)[1] = *(const uint4*)(b1p + 16);
      ((uint4*)&b2)[0] = *(const uint4*)(b2p);
      ((uint4*)&b2)[1] = *(const uint4*)(b2p + 16);
      ((uint4*)&b3)[0] = *(const uint4*)(b3p);
      ((uint4*)&b3)[1] = *(const uint4*)(b3p + 16);

      acc0 = __builtin_amdgcn_wmma_f32_16x16x32_bf16(false, a, false, b0,
                                                     (short)0, acc0, false, false);
      acc1 = __builtin_amdgcn_wmma_f32_16x16x32_bf16(false, a, false, b1,
                                                     (short)0, acc1, false, false);
      acc2 = __builtin_amdgcn_wmma_f32_16x16x32_bf16(false, a, false, b2,
                                                     (short)0, acc2, false, false);
      acc3 = __builtin_amdgcn_wmma_f32_16x16x32_bf16(false, a, false, b3,
                                                     (short)0, acc3, false, false);
    }

    // Ensure next chunk's DMA landed before anyone reads that buffer,
    // and that everyone is done with this buffer before it is re-targeted.
    if (wv == 0 && (cchunk + 1) < NCHUNK) {
      __builtin_amdgcn_s_wait_tensorcnt(0);
    }
    __syncthreads();
  }

  // C/D layout: VGPR j holds (M = j + 8*(lane/16), N = lane%16).
  const int hcol = nBase + nLoc;
  const float bi = bias[0 * H_SZ + hcol];
  const float bf = bias[1 * H_SZ + hcol];
  const float bg = bias[2 * H_SZ + hcol];
  const float bo = bias[3 * H_SZ + hcol];
  const int mHi = (lane >> 4) << 3;

#pragma unroll
  for (int j = 0; j < 8; ++j) {
    const int m = mBase + j + mHi;               // batch row
    const size_t sidx = (size_t)m * H_SZ + hcol; // state index
    float iv = sigmoid_f(acc0[j] + bi);
    float fv = sigmoid_f(acc1[j] + bf);
    float gv = tanh_f   (acc2[j] + bg);
    float ov = sigmoid_f(acc3[j] + bo);
    float cN = fv * c[sidx] + iv * gv;
    float hN = ov * tanh_f(cN);
    c[sidx] = cN;
    hout[sidx] = f32_to_bf16_rn(hN);
    // hs is a 128MB streamed output nobody re-reads: keep it out of L2
    // so the 16MB bf16 weight matrix stays resident across all 512 steps.
    __builtin_nontemporal_store(hN, &out[((size_t)t * B_SZ + m) * H_SZ + hcol]);
    if (t == T_STEPS - 1) {
      const size_t hsEnd = (size_t)T_STEPS * B_SZ * H_SZ;
      out[hsEnd + sidx] = hN;                          // hT
      out[hsEnd + (size_t)B_SZ * H_SZ + sidx] = cN;    // cT
    }
  }
}

extern "C" void kernel_launch(void* const* d_in, const int* in_sizes, int n_in,
                              void* d_out, int out_size, void* d_ws, size_t ws_size,
                              hipStream_t stream) {
  (void)in_sizes; (void)n_in; (void)out_size; (void)ws_size;
  const float* x    = (const float*)d_in[0];
  const float* Wx   = (const float*)d_in[1];
  const float* Wh   = (const float*)d_in[2];
  const float* bias = (const float*)d_in[3];
  float* out = (float*)d_out;

  // Workspace layout (16B-aligned offsets), ~84.4 MB total:
  //   Wt  : bf16 [4096, 2048]  = 16 MB
  //   xbf : bf16 [512,64,1024] = 64 MB
  //   hA/hB : bf16 [64,1024]   = 2 x 128 KB
  //   c   : f32  [64,1024]     = 256 KB
  char* ws = (char*)d_ws;
  const size_t wtBytes = (size_t)N_G * K_TOT * sizeof(unsigned short);
  const size_t xbBytes = (size_t)T_STEPS * B_SZ * IN_SZ * sizeof(unsigned short);
  const size_t hBytes  = (size_t)B_SZ * H_SZ * sizeof(unsigned short);
  unsigned short* Wt  = (unsigned short*)ws;
  unsigned short* xbf = (unsigned short*)(ws + wtBytes);
  unsigned short* hA  = (unsigned short*)(ws + wtBytes + xbBytes);
  unsigned short* hB  = (unsigned short*)(ws + wtBytes + xbBytes + hBytes);
  float*          c   = (float*)(ws + wtBytes + xbBytes + 2 * hBytes);

  {
    const size_t nW = (size_t)N_G * K_TOT;
    convert_w_kernel<<<(unsigned)((nW + 255) / 256), 256, 0, stream>>>(Wx, Wh, Wt);
    const size_t nX = (size_t)T_STEPS * B_SZ * IN_SZ;
    convert_x_kernel<<<(unsigned)((nX + 255) / 256), 256, 0, stream>>>(x, xbf, nX);
    init_state_kernel<<<(B_SZ * H_SZ + 255) / 256, 256, 0, stream>>>(hA, c);
  }

  // Sequential recurrence: one launch per step = device-wide barrier.
  // h is double-buffered (read hin, write hout) to avoid cross-block races.
  for (int t = 0; t < T_STEPS; ++t) {
    const unsigned short* hin = (t & 1) ? hB : hA;
    unsigned short* hout      = (t & 1) ? hA : hB;
    lstm_step_kernel<<<64, 128, 0, stream>>>(xbf, Wt, bias, hin, hout, c, out, t);
  }
}
